// LSTM_GNN_Model_3702261809802
// MI455X (gfx1250) — compile-verified, matrix operand order checked by reading, not run on hardware
//
#include <hip/hip_runtime.h>
#include <hip/hip_bf16.h>

// ---------------------------------------------------------------------------
// Problem constants (from the reference)
// ---------------------------------------------------------------------------
#define BB   16
#define SS   1024
#define DIN  128
#define HH   256           // hidden
#define GG   256           // GCN width
#define CC   17            // classes
#define NN   (BB * SS)     // 16384 nodes
#define G4H  (4 * HH)      // 1024 gate columns

typedef float  v2f  __attribute__((ext_vector_type(2)));
typedef float  v8f  __attribute__((ext_vector_type(8)));
typedef __bf16 v8bf __attribute__((ext_vector_type(8)));
typedef __bf16 v16bf __attribute__((ext_vector_type(16)));

// fp32 WMMA 16x16x4  (D = A(16x4) * B(4x16) + C), wave32
static __device__ inline v8f wmma_f32_k4(v2f a, v2f b, v8f c) {
#if __has_builtin(__builtin_amdgcn_wmma_f32_16x16x4_f32)
    return __builtin_amdgcn_wmma_f32_16x16x4_f32(false, a, false, b, (short)0, c,
                                                 false, false);
#else
    c[0] += a[0] * b[0] + a[1] * b[1];   // keep-compiling fallback
    return c;
#endif
}

// ---------------------------------------------------------------------------
// K1: xg[t][b][g] = sum_k x[b][t][k] * w_ih[g][k] + (b_ih[g]+b_hh[g])
// grid (8, S), block 256 (8 waves -> 128 gate columns / block)
// ---------------------------------------------------------------------------
__global__ __launch_bounds__(256)
void xg_gemm(const float* __restrict__ x, const float* __restrict__ w_ih,
             const float* __restrict__ b_ih, const float* __restrict__ b_hh,
             float* __restrict__ xg)
{
    const int tid    = threadIdx.x;
    const int lane15 = tid & 15;
    const int half   = (tid >> 4) & 1;
    const int wave   = tid >> 5;
    const int t      = blockIdx.y;
    const int gcol   = blockIdx.x * 128 + wave * 16 + lane15;

    v8f acc = {};
    const float* arow = x + ((size_t)lane15 * SS + t) * DIN;   // A: M=batch
    const float* brow = w_ih + (size_t)gcol * DIN;             // B: N=gate col
#pragma unroll
    for (int k0 = 0; k0 < DIN; k0 += 4) {
        v2f a = *(const v2f*)(arow + k0 + 2 * half);
        v2f b = *(const v2f*)(brow + k0 + 2 * half);
        acc = wmma_f32_k4(a, b, acc);
    }
    const float bias = b_ih[gcol] + b_hh[gcol];
    float* orow = xg + (size_t)t * BB * G4H;
#pragma unroll
    for (int r = 0; r < 8; ++r)                 // D: M = r + 8*half, N = gcol
        orow[(size_t)(r + 8 * half) * G4H + gcol] = acc[r] + bias;
}

// ---------------------------------------------------------------------------
// K2: persistent LSTM recurrence, one workgroup (32 waves) per direction.
//     gates = xg[t] + h @ w_hh^T via WMMA f32 16x16x4; h in LDS, c in regs.
//     xg[t+1] slab is double-buffered into LDS with async global->LDS copies
//     (GLOBAL_LOAD_ASYNC_TO_LDS_B128, retired with s_wait_asynccnt) so the
//     serial critical path never waits on global memory for the C-init.
// grid 2, block 1024, dynamic LDS (~208 KB of the WGP's 320 KB)
// ---------------------------------------------------------------------------
#define H_STR 258   // padded h row stride (floats) to avoid LDS bank conflicts
#define XSLAB (BB * G4H)            // 16384 floats per time-step slab

__global__ __launch_bounds__(1024)
void lstm_rec(const float* __restrict__ xg_f, const float* __restrict__ xg_r,
              const float* __restrict__ whh_f, const float* __restrict__ whh_r,
              __bf16* __restrict__ nodes_bf)
{
    extern __shared__ float smem[];
    float* h_sh     = smem;                        // 16 * H_STR
    float* gates_sh = smem + BB * H_STR;           // 16 * 1024
    float* xbuf[2]  = { gates_sh + XSLAB, gates_sh + 2 * XSLAB };

    // LDS byte offsets of the two xg staging buffers (for async-load VDST)
    const unsigned lds0 = __builtin_amdgcn_groupstaticsize();
    const unsigned xoff[2] = { lds0 + (unsigned)(BB * H_STR + XSLAB) * 4u,
                               lds0 + (unsigned)(BB * H_STR + 2 * XSLAB) * 4u };

    const int dir = blockIdx.x;
    const float* xg  = dir ? xg_r  : xg_f;
    const float* whh = dir ? whh_r : whh_f;

    const int tid    = threadIdx.x;
    const int lane15 = tid & 15;
    const int half   = (tid >> 4) & 1;
    const int wave   = tid >> 5;

    for (int i = tid; i < BB * H_STR; i += 1024) h_sh[i] = 0.f;

    // ---- preload xg[t0] into buffer 0: 16 floats (4 x b128) per thread ----
    {
        const int t0 = dir ? (SS - 1) : 0;
        const float* gsrc = xg + (size_t)t0 * XSLAB + tid * 16;
        const unsigned loff = xoff[0] + (unsigned)tid * 64u;
#pragma unroll
        for (int i = 0; i < 4; ++i)
            asm volatile("global_load_async_to_lds_b128 %0, %1, off"
                         :: "v"(loff + i * 16u), "v"(gsrc + i * 4) : "memory");
    }
    asm volatile("s_wait_asynccnt 0x0" ::: "memory");
    __syncthreads();

    float c_reg[4] = {0.f, 0.f, 0.f, 0.f};
    const int e0 = tid * 4;                  // this thread's 4 (b,j) cells

    for (int ti = 0; ti < SS; ++ti) {
        const int t = dir ? (SS - 1 - ti) : ti;
        const float* xcur = xbuf[ti & 1];

        // ---- kick off async staging of xg[t+1] into the other buffer ----
        if (ti + 1 < SS) {
            const int tn = dir ? (SS - 2 - ti) : (ti + 1);
            const float* gsrc = xg + (size_t)tn * XSLAB + tid * 16;
            const unsigned loff = xoff[(ti + 1) & 1] + (unsigned)tid * 64u;
#pragma unroll
            for (int i = 0; i < 4; ++i)
                asm volatile("global_load_async_to_lds_b128 %0, %1, off"
                             :: "v"(loff + i * 16u), "v"(gsrc + i * 4) : "memory");
        }

        // ---- gates = xg[t] + h @ w_hh^T : 2 N-tiles of 16 cols per wave ----
#pragma unroll
        for (int tile = 0; tile < 2; ++tile) {
            const int gcol = wave * 32 + tile * 16 + lane15;
            v8f acc;
#pragma unroll
            for (int r = 0; r < 8; ++r)       // C init from LDS-staged slab
                acc[r] = xcur[(r + 8 * half) * G4H + gcol];
            const float* arow = h_sh + lane15 * H_STR;       // A: M=batch (LDS)
            const float* brow = whh + (size_t)gcol * HH;     // B: N=gate col (L2)
#pragma unroll 4
            for (int k0 = 0; k0 < HH; k0 += 4) {
                v2f a = *(const v2f*)(arow + k0 + 2 * half);
                v2f b = *(const v2f*)(brow + k0 + 2 * half);
                acc = wmma_f32_k4(a, b, acc);
            }
#pragma unroll
            for (int r = 0; r < 8; ++r)
                gates_sh[(r + 8 * half) * G4H + gcol] = acc[r];
        }
        __syncthreads();

        // ---- elementwise gate math: i,f,g,o ; update c (regs) and h (LDS) ----
#pragma unroll
        for (int i = 0; i < 4; ++i) {
            const int e = e0 + i;
            const int b = e >> 8;
            const int j = e & 255;
            const float* gb = gates_sh + b * G4H;
            float ig = gb[j], fg = gb[HH + j], gg = gb[2 * HH + j], og = gb[3 * HH + j];
            ig = 1.f / (1.f + __expf(-ig));
            fg = 1.f / (1.f + __expf(-fg));
            og = 1.f / (1.f + __expf(-og));
            gg = tanhf(gg);
            const float c = fg * c_reg[i] + ig * gg;
            c_reg[i] = c;
            const float h = og * tanhf(c);
            h_sh[b * H_STR + j] = h;
            nodes_bf[(size_t)(b * SS + t) * (2 * HH) + dir * HH + j] = (__bf16)h;
        }
        // next buffer must be resident before anyone reads it next iteration
        asm volatile("s_wait_asynccnt 0x0" ::: "memory");
        __syncthreads();
    }
}

// ---------------------------------------------------------------------------
// K3: weight transpose + bf16 convert: WT[n][k] = bf16(W[k][n])
// ---------------------------------------------------------------------------
__global__ void transpose_bf16(const float* __restrict__ W, __bf16* __restrict__ WT,
                               int K, int Nc)
{
    const int id = blockIdx.x * 256 + threadIdx.x;
    if (id >= K * Nc) return;
    const int k = id / Nc, n = id % Nc;
    WT[(size_t)n * K + k] = (__bf16)W[id];
}

// ---------------------------------------------------------------------------
// K4: bf16 GEMM  C[M,256] = A[M,K] (bf16, row-major) x BT[256,K] (bf16, n-major)
//     V_WMMA_F32_16X16X32_BF16 ; grid (2, M/16), block 256 (8 waves)
// ---------------------------------------------------------------------------
__global__ __launch_bounds__(256)
void gemm_bf16(const __bf16* __restrict__ A, const __bf16* __restrict__ BT,
               float* __restrict__ C, int K)
{
    const int tid    = threadIdx.x;
    const int lane15 = tid & 15;
    const int half   = (tid >> 4) & 1;
    const int wave   = tid >> 5;
    const int m0     = blockIdx.y * 16;
    const int n0     = (blockIdx.x * 8 + wave) * 16;

    v8f acc = {};
    const __bf16* arow = A  + (size_t)(m0 + lane15) * K;
    const __bf16* brow = BT + (size_t)(n0 + lane15) * K;
    for (int k0 = 0; k0 < K; k0 += 32) {
        v8bf alo = *(const v8bf*)(arow + k0 + 8 * half);
        v8bf ahi = *(const v8bf*)(arow + k0 + 16 + 8 * half);
        v8bf blo = *(const v8bf*)(brow + k0 + 8 * half);
        v8bf bhi = *(const v8bf*)(brow + k0 + 16 + 8 * half);
        v16bf av = __builtin_shufflevector(alo, ahi, 0,1,2,3,4,5,6,7,8,9,10,11,12,13,14,15);
        v16bf bv = __builtin_shufflevector(blo, bhi, 0,1,2,3,4,5,6,7,8,9,10,11,12,13,14,15);
        acc = __builtin_amdgcn_wmma_f32_16x16x32_bf16(false, av, false, bv, (short)0,
                                                      acc, false, false);
    }
#pragma unroll
    for (int r = 0; r < 8; ++r)
        C[(size_t)(m0 + r + 8 * half) * GG + n0 + lane15] = acc[r];
}

// ---------------------------------------------------------------------------
// K5: GCN degree / norm / scatter helpers
// ---------------------------------------------------------------------------
__global__ void deg_init(float* __restrict__ deg)
{   deg[blockIdx.x * 256 + threadIdx.x] = 1.0f; }           // self-loop

__global__ void deg_count(const long long* __restrict__ dst, float* __restrict__ deg, int E)
{
    const int e = blockIdx.x * 256 + threadIdx.x;
    if (e < E) atomicAdd(&deg[dst[e]], 1.0f);
}

__global__ void deg_inv(const float* __restrict__ deg, float* __restrict__ inv)
{
    const int n = blockIdx.x * 256 + threadIdx.x;
    inv[n] = rsqrtf(fmaxf(deg[n], 1e-12f));
}

// acc[n][g] = bias[g] + h[n][g] * inv[n]^2   (self-loop message)
__global__ void gcn_self(const float* __restrict__ h, const float* __restrict__ inv,
                         const float* __restrict__ bias, float* __restrict__ acc)
{
    const int id = blockIdx.x * 256 + threadIdx.x;      // N*G threads
    const int n = id >> 8, g = id & 255;
    const float iv = inv[n];
    acc[id] = bias[g] + h[id] * iv * iv;
}

// acc[dst] += h[src] * inv[src]*inv[dst] ; 4 features per thread
__global__ void gcn_scatter(const float* __restrict__ h, const long long* __restrict__ src,
                            const long long* __restrict__ dst, const float* __restrict__ inv,
                            float* __restrict__ acc, int E)
{
    const int id = blockIdx.x * 256 + threadIdx.x;
    const int e = id >> 6;
    if (e >= E) return;
    const int g = (id & 63) * 4;
    const long long s = src[e], d = dst[e];
    const float w = inv[s] * inv[d];
    const float4 v = *(const float4*)(h + (size_t)s * GG + g);
    float* ap = acc + (size_t)d * GG + g;
    atomicAdd(ap + 0, v.x * w);
    atomicAdd(ap + 1, v.y * w);
    atomicAdd(ap + 2, v.z * w);
    atomicAdd(ap + 3, v.w * w);
}

__global__ void relu_to_bf16(const float* __restrict__ acc, __bf16* __restrict__ out)
{
    const int id = blockIdx.x * 256 + threadIdx.x;
    out[id] = (__bf16)fmaxf(acc[id], 0.f);
}

// ---------------------------------------------------------------------------
// K6: classifier  out[n][c] = bc[c] + sum_k relu(g2[n][k]) * Wc[k][c]   (C=17)
// ---------------------------------------------------------------------------
__global__ void classify(const float* __restrict__ g2, const float* __restrict__ Wc,
                         const float* __restrict__ bc, float* __restrict__ out)
{
    const int id = blockIdx.x * 256 + threadIdx.x;
    if (id >= NN * CC) return;
    const int n = id / CC, c = id % CC;
    const float* row = g2 + (size_t)n * GG;
    float s = bc[c];
    for (int k = 0; k < GG; ++k)
        s += fmaxf(row[k], 0.f) * Wc[k * CC + c];
    out[id] = s;
}

// ---------------------------------------------------------------------------
extern "C" void kernel_launch(void* const* d_in, const int* in_sizes, int n_in,
                              void* d_out, int out_size, void* d_ws, size_t ws_size,
                              hipStream_t stream)
{
    (void)n_in; (void)out_size; (void)ws_size;
    const float*     x      = (const float*)d_in[0];
    const long long* ei     = (const long long*)d_in[1];   // [2,E] int64
    const float*     w_ih_f = (const float*)d_in[2];
    const float*     w_hh_f = (const float*)d_in[3];
    const float*     b_ih_f = (const float*)d_in[4];
    const float*     b_hh_f = (const float*)d_in[5];
    const float*     w_ih_r = (const float*)d_in[6];
    const float*     w_hh_r = (const float*)d_in[7];
    const float*     b_ih_r = (const float*)d_in[8];
    const float*     b_hh_r = (const float*)d_in[9];
    const float*     W1     = (const float*)d_in[10];
    const float*     b1     = (const float*)d_in[11];
    const float*     W2     = (const float*)d_in[12];
    const float*     b2     = (const float*)d_in[13];
    const float*     Wc     = (const float*)d_in[14];
    const float*     bc     = (const float*)d_in[15];
    float* out = (float*)d_out;
    const int E = in_sizes[1] / 2;

    // ---- workspace layout (xg region is reused by the GCN f32 buffers) ----
    char* ws = (char*)d_ws;
    constexpr size_t MB = 1024ull * 1024ull;
    float*  xg_f = (float*)(ws + 0);            // 64 MB, dead after lstm_rec
    float*  xg_r = (float*)(ws + 64 * MB);      // 64 MB, dead after lstm_rec
    float*  h1   = (float*)(ws + 0);            // overlay
    float*  acc1 = (float*)(ws + 16 * MB);      // overlay
    float*  h2   = (float*)(ws + 32 * MB);      // overlay
    float*  acc2 = (float*)(ws + 48 * MB);      // overlay
    __bf16* nodes_bf = (__bf16*)(ws + 128 * MB); // N*512 bf16 = 16 MB
    __bf16* g1bf     = (__bf16*)(ws + 144 * MB); // N*256 bf16 =  8 MB
    float*  deg      = (float*)(ws + 152 * MB);
    float*  inv      = (float*)(ws + 152 * MB + 64 * 1024);
    __bf16* W1T      = (__bf16*)(ws + 153 * MB);             // 256 KB
    __bf16* W2T      = (__bf16*)(ws + 153 * MB + 512 * 1024);// 128 KB

    const long long* esrc = ei;
    const long long* edst = ei + E;

    // 1) precompute input gate contributions (fp32 WMMA)
    xg_gemm<<<dim3(8, SS), 256, 0, stream>>>(x, w_ih_f, b_ih_f, b_hh_f, xg_f);
    xg_gemm<<<dim3(8, SS), 256, 0, stream>>>(x, w_ih_r, b_ih_r, b_hh_r, xg_r);

    // 2) weights -> bf16, n-major
    transpose_bf16<<<(2 * HH * GG + 255) / 256, 256, 0, stream>>>(W1, W1T, 2 * HH, GG);
    transpose_bf16<<<(GG * GG + 255) / 256, 256, 0, stream>>>(W2, W2T, GG, GG);

    // 3) degree / symmetric norm
    deg_init<<<NN / 256, 256, 0, stream>>>(deg);
    deg_count<<<(E + 255) / 256, 256, 0, stream>>>(edst, deg, E);
    deg_inv<<<NN / 256, 256, 0, stream>>>(deg, inv);

    // 4) bidirectional LSTM recurrence (persistent, fp32 WMMA + async LDS staging)
    const size_t smem = (size_t)(BB * H_STR + 3 * XSLAB) * sizeof(float); // ~208 KB
    lstm_rec<<<2, 1024, smem, stream>>>(xg_f, xg_r, w_hh_f, w_hh_r, nodes_bf);

    // 5) GCN layer 1: bf16 GEMM (K=512) + norm-scatter + relu
    gemm_bf16<<<dim3(2, NN / 16), 256, 0, stream>>>(nodes_bf, W1T, h1, 2 * HH);
    gcn_self<<<NN * GG / 256, 256, 0, stream>>>(h1, inv, b1, acc1);
    gcn_scatter<<<((size_t)E * 64 + 255) / 256, 256, 0, stream>>>(h1, esrc, edst, inv, acc1, E);
    relu_to_bf16<<<NN * GG / 256, 256, 0, stream>>>(acc1, g1bf);

    // 6) GCN layer 2: bf16 GEMM (K=256) + norm-scatter (relu fused in classifier)
    gemm_bf16<<<dim3(2, NN / 16), 256, 0, stream>>>(g1bf, W2T, h2, GG);
    gcn_self<<<NN * GG / 256, 256, 0, stream>>>(h2, inv, b2, acc2);
    gcn_scatter<<<((size_t)E * 64 + 255) / 256, 256, 0, stream>>>(h2, esrc, edst, inv, acc2, E);

    // 7) classifier
    classify<<<(NN * CC + 255) / 256, 256, 0, stream>>>(acc2, Wc, bc, out);
}
